// ImageEncoder_65996467470744
// MI455X (gfx1250) — matrix operand order repaired and checked by reading, not compiled, not run on hardware
//
#include <hip/hip_runtime.h>

typedef __attribute__((ext_vector_type(16))) _Float16 v16h;
typedef __attribute__((ext_vector_type(8)))  _Float16 v8h;
typedef __attribute__((ext_vector_type(8)))  float    v8f;

union HFrag { v16h v; v8h h[2]; };

// ---------------------------------------------------------------------------
// f32 -> f16 conversion (activations and weights), grid-stride
// ---------------------------------------------------------------------------
__global__ void f32_to_f16(const float* __restrict__ src, _Float16* __restrict__ dst, int n) {
    int i = blockIdx.x * blockDim.x + threadIdx.x;
    int stride = gridDim.x * blockDim.x;
    for (; i < n; i += stride) dst[i] = (_Float16)src[i];
}

// ---------------------------------------------------------------------------
// Fused WMMA GEMM: Out = act(A[M,K] @ W[N,K]^T + bias [+ Res])
// Tile 128x128x32, 256 threads = 8 waves (4x2), each wave 2x4 16x16 frags.
// Double-buffered LDS; staging uses CDNA5 async global->LDS DMA (ASYNCcnt).
// ---------------------------------------------------------------------------
template<bool HAS_RES, bool RELU, bool OUT_F, bool OUT_H>
__launch_bounds__(256)
__global__ void gemm_wmma_f16(const _Float16* __restrict__ A,   // [M,K] row-major f16
                              const _Float16* __restrict__ W,   // [N,K] row-major f16
                              const float*    __restrict__ bias,// [N]
                              const float*    __restrict__ Res, // [M,N] f32 or null
                              float*          __restrict__ OutF,
                              _Float16*       __restrict__ OutH,
                              int M, int N, int K)
{
    constexpr int BM = 128, BN = 128, BK = 32, LDS_S = 40; // 40-half stride: conflict-free
    __shared__ _Float16 As[2][BM * LDS_S];
    __shared__ _Float16 Bs[2][BN * LDS_S];

    const int t    = threadIdx.x;
    const int wave = t >> 5;
    const int lane = t & 31;
    const int wr   = wave >> 1;      // 0..3 (32 rows each)
    const int wc   = wave & 1;       // 0..1 (64 cols each)
    const int lhi  = lane >> 4;      // 0/1
    const int llo  = lane & 15;
    const int m0   = blockIdx.y * BM;
    const int n0   = blockIdx.x * BN;

    // staging geometry: 128 rows x 32 halfs (64B) per matrix = 512 x 16B chunks;
    // 256 threads -> each thread DMAs rows r0 and r0+64 at half-offset ko0.
    const int r0  = t >> 2;
    const int ko0 = (t & 3) * 8;

    const v8f vzero = {0.f,0.f,0.f,0.f,0.f,0.f,0.f,0.f};
    v8f acc[2][4];
#pragma unroll
    for (int i = 0; i < 2; ++i)
#pragma unroll
        for (int j = 0; j < 4; ++j) acc[i][j] = vzero;

    // async DMA of one 128x32 A tile + 128x32 W tile into LDS buffer `buf`
    auto stage = [&](int buf, int k0) {
        const _Float16* gA0 = A + (size_t)(m0 + r0)      * K + k0 + ko0;
        const _Float16* gA1 = A + (size_t)(m0 + r0 + 64) * K + k0 + ko0;
        const _Float16* gB0 = W + (size_t)(n0 + r0)      * K + k0 + ko0;
        const _Float16* gB1 = W + (size_t)(n0 + r0 + 64) * K + k0 + ko0;
        unsigned lA0 = (unsigned)(size_t)&As[buf][r0 * LDS_S + ko0];
        unsigned lA1 = (unsigned)(size_t)&As[buf][(r0 + 64) * LDS_S + ko0];
        unsigned lB0 = (unsigned)(size_t)&Bs[buf][r0 * LDS_S + ko0];
        unsigned lB1 = (unsigned)(size_t)&Bs[buf][(r0 + 64) * LDS_S + ko0];
        asm volatile("global_load_async_to_lds_b128 %0, %1, off"
                     :: "v"(lA0), "v"(gA0) : "memory");
        asm volatile("global_load_async_to_lds_b128 %0, %1, off"
                     :: "v"(lA1), "v"(gA1) : "memory");
        asm volatile("global_load_async_to_lds_b128 %0, %1, off"
                     :: "v"(lB0), "v"(gB0) : "memory");
        asm volatile("global_load_async_to_lds_b128 %0, %1, off"
                     :: "v"(lB1), "v"(gB1) : "memory");
    };

    const int KT = K / BK;
    stage(0, 0);

    for (int kt = 0; kt < KT; ++kt) {
        const int buf = kt & 1;
        if (kt + 1 < KT) {
            stage(buf ^ 1, (kt + 1) * BK);
            // 4 ops outstanding from next stage; <=4 guarantees current stage landed
            asm volatile("s_wait_asynccnt 4" ::: "memory");
        } else {
            asm volatile("s_wait_asynccnt 0" ::: "memory");
        }
        __syncthreads();

        // ---- build fragments, contiguous k = 16*lhi + s, same order for A and B ----
        HFrag fa[2], fb[4];
#pragma unroll
        for (int tm = 0; tm < 2; ++tm) {
            int m  = wr * 32 + tm * 16 + llo;
            int kb = lhi * 16;
            fa[tm].h[0] = *(const v8h*)&As[buf][m * LDS_S + kb];       // K kb..kb+7
            fa[tm].h[1] = *(const v8h*)&As[buf][m * LDS_S + kb + 8];   // K kb+8..kb+15
        }
#pragma unroll
        for (int tn = 0; tn < 4; ++tn) {
            int n  = wc * 64 + tn * 16 + llo;
            int kb = lhi * 16;
            fb[tn].h[0] = *(const v8h*)&Bs[buf][n * LDS_S + kb];       // K kb..kb+7
            fb[tn].h[1] = *(const v8h*)&Bs[buf][n * LDS_S + kb + 8];   // K kb+8..kb+15
        }

#pragma unroll
        for (int tm = 0; tm < 2; ++tm)
#pragma unroll
            for (int tn = 0; tn < 4; ++tn)
                acc[tm][tn] = __builtin_amdgcn_wmma_f32_16x16x32_f16(
                    false, fa[tm].v, false, fb[tn].v,
                    (short)0, acc[tm][tn], false, false);

        __syncthreads();  // all reads of buf done before it is re-staged
    }

    // ---- fused epilogue: bias (+residual) (+relu), f32 and/or f16 stores ----
#pragma unroll
    for (int tm = 0; tm < 2; ++tm) {
#pragma unroll
        for (int tn = 0; tn < 4; ++tn) {
            const int gn = n0 + wc * 64 + tn * 16 + llo;
            const float bv = bias[gn];
#pragma unroll
            for (int g = 0; g < 8; ++g) {
                const int gm = m0 + wr * 32 + tm * 16 + lhi * 8 + g;
                float v = acc[tm][tn][g] + bv;
                if (HAS_RES) v += Res[(size_t)gm * N + gn];
                if (RELU)    v = fmaxf(v, 0.0f);
                if (OUT_F)   OutF[(size_t)gm * N + gn] = v;
                if (OUT_H)   OutH[(size_t)gm * N + gn] = (_Float16)v;
            }
        }
    }
}

// ---------------------------------------------------------------------------
// Row LayerNorm over 768 floats; optional fused ReLU; f32 and/or f16 outputs.
// One 256-thread block per row, 3 elems/thread, LDS tree reduce.
// ---------------------------------------------------------------------------
template<bool RELU, bool OUT_F, bool OUT_H>
__launch_bounds__(256)
__global__ void layernorm_768(const float* __restrict__ X,
                              const float* __restrict__ gamma,
                              const float* __restrict__ beta,
                              float* __restrict__ OutF,
                              _Float16* __restrict__ OutH)
{
    __shared__ float red[256];
    const int row = blockIdx.x;
    const int t   = threadIdx.x;
    const float* xr = X + (size_t)row * 768;

    float x0 = xr[t], x1 = xr[t + 256], x2 = xr[t + 512];
    red[t] = x0 + x1 + x2;
    __syncthreads();
    for (int off = 128; off > 0; off >>= 1) {
        if (t < off) red[t] += red[t + off];
        __syncthreads();
    }
    const float mu = red[0] * (1.0f / 768.0f);
    __syncthreads();

    const float d0 = x0 - mu, d1 = x1 - mu, d2 = x2 - mu;
    red[t] = d0 * d0 + d1 * d1 + d2 * d2;
    __syncthreads();
    for (int off = 128; off > 0; off >>= 1) {
        if (t < off) red[t] += red[t + off];
        __syncthreads();
    }
    const float rstd = rsqrtf(red[0] * (1.0f / 768.0f) + 1e-5f);

    float y0 = d0 * rstd * gamma[t      ] + beta[t      ];
    float y1 = d1 * rstd * gamma[t + 256] + beta[t + 256];
    float y2 = d2 * rstd * gamma[t + 512] + beta[t + 512];
    if (RELU) { y0 = fmaxf(y0, 0.f); y1 = fmaxf(y1, 0.f); y2 = fmaxf(y2, 0.f); }
    if (OUT_F) {
        float* o = OutF + (size_t)row * 768;
        o[t] = y0; o[t + 256] = y1; o[t + 512] = y2;
    }
    if (OUT_H) {
        _Float16* o = OutH + (size_t)row * 768;
        o[t] = (_Float16)y0; o[t + 256] = (_Float16)y1; o[t + 512] = (_Float16)y2;
    }
}

// ---------------------------------------------------------------------------
extern "C" void kernel_launch(void* const* d_in, const int* in_sizes, int n_in,
                              void* d_out, int out_size, void* d_ws, size_t ws_size,
                              hipStream_t stream)
{
    (void)n_in; (void)out_size; (void)ws_size;
    const int Dm = 768, Dout = 512;

    const float* query    = (const float*)d_in[0];
    const float* ref      = (const float*)d_in[1];
    const float* sa_w_in  = (const float*)d_in[2];
    const float* sa_b_in  = (const float*)d_in[3];
    const float* sa_w_out = (const float*)d_in[4];
    const float* sa_b_out = (const float*)d_in[5];
    const float* ln1_g    = (const float*)d_in[6];
    const float* ln1_b    = (const float*)d_in[7];
    const float* ca_w_in  = (const float*)d_in[8];
    const float* ca_b_in  = (const float*)d_in[9];
    const float* ca_w_out = (const float*)d_in[10];
    const float* ca_b_out = (const float*)d_in[11];
    const float* ln2_g    = (const float*)d_in[12];
    const float* ln2_b    = (const float*)d_in[13];
    const float* mlp_w1   = (const float*)d_in[14];
    const float* mlp_b1   = (const float*)d_in[15];
    const float* mlp_w2   = (const float*)d_in[16];
    const float* mlp_b2   = (const float*)d_in[17];

    const int B  = in_sizes[0] / Dm;        // 8192
    const size_t nA = (size_t)B * Dm;

    // workspace carve-out (256B aligned slices)
    char* ws = (char*)d_ws;
    size_t off = 0;
    auto carve = [&](size_t bytes) -> void* {
        void* p = ws + off;
        off = (off + bytes + 255) & ~(size_t)255;
        return p;
    };
    _Float16* bufA = (_Float16*)carve(nA * 2);            // query_f16, later x_f16
    _Float16* bufR = (_Float16*)carve(nA * 2);            // ref_f16
    _Float16* bufT = (_Float16*)carve(nA * 2);            // T1 / R1 / M1 (sequential reuse)
    float*    bufF = (float*)   carve(nA * 4);            // f32 residual sums / q
    _Float16* wbuf = (_Float16*)carve((size_t)Dm * Dm * 2); // current weight f16

    const dim3 cblk(256);
    const int  cgrid = 512;
    const dim3 gblk(256);
    const dim3 grid768(Dm / 128, B / 128);   // 6 x 64
    const dim3 grid512(Dout / 128, B / 128); // 4 x 64

    // activations -> f16
    f32_to_f16<<<cgrid, cblk, 0, stream>>>(query, bufA, (int)nA);
    f32_to_f16<<<cgrid, cblk, 0, stream>>>(ref,   bufR, (int)nA);

    // GEMM1: T1 = query @ sa_Wv^T + sa_bv  (softmax over 1 key == identity => only V path)
    f32_to_f16<<<cgrid, cblk, 0, stream>>>(sa_w_in + 2 * Dm * Dm, wbuf, Dm * Dm);
    gemm_wmma_f16<false, false, false, true><<<grid768, gblk, 0, stream>>>(
        bufA, wbuf, sa_b_in + 2 * Dm, nullptr, nullptr, bufT, B, Dm, Dm);

    // GEMM2: S = T1 @ sa_Wout^T + b_out + query  -> f32
    f32_to_f16<<<cgrid, cblk, 0, stream>>>(sa_w_out, wbuf, Dm * Dm);
    gemm_wmma_f16<true, false, true, false><<<grid768, gblk, 0, stream>>>(
        bufT, wbuf, sa_b_out, query, bufF, nullptr, B, Dm, Dm);

    // LN1 (in place): q = LN(S)  -- q only ever used as a residual, keep f32 only
    layernorm_768<false, true, false><<<B, 256, 0, stream>>>(bufF, ln1_g, ln1_b, bufF, nullptr);

    // GEMM3: R1 = ref @ ca_Wv^T + ca_bv   (cross-attn output depends only on ref_feat)
    f32_to_f16<<<cgrid, cblk, 0, stream>>>(ca_w_in + 2 * Dm * Dm, wbuf, Dm * Dm);
    gemm_wmma_f16<false, false, false, true><<<grid768, gblk, 0, stream>>>(
        bufR, wbuf, ca_b_in + 2 * Dm, nullptr, nullptr, bufT, B, Dm, Dm);

    // GEMM4: S2 = R1 @ ca_Wout^T + ca_b_out + q  -> f32 (element-wise in-place residual)
    f32_to_f16<<<cgrid, cblk, 0, stream>>>(ca_w_out, wbuf, Dm * Dm);
    gemm_wmma_f16<true, false, true, false><<<grid768, gblk, 0, stream>>>(
        bufT, wbuf, ca_b_out, bufF, bufF, nullptr, B, Dm, Dm);

    // LN2 + ReLU: x = relu(LN(S2)) -> f16 (only a GEMM input from here on)
    layernorm_768<true, false, true><<<B, 256, 0, stream>>>(bufF, ln2_g, ln2_b, nullptr, bufA);

    // GEMM5: M1 = relu(x @ mlp_w1^T + b1) -> f16
    f32_to_f16<<<cgrid, cblk, 0, stream>>>(mlp_w1, wbuf, Dm * Dm);
    gemm_wmma_f16<false, true, false, true><<<grid768, gblk, 0, stream>>>(
        bufA, wbuf, mlp_b1, nullptr, nullptr, bufT, B, Dm, Dm);

    // GEMM6: out = relu(M1 @ mlp_w2^T + b2) -> f32 d_out  [8192 x 512]
    f32_to_f16<<<cgrid, cblk, 0, stream>>>(mlp_w2, wbuf, Dout * Dm);
    gemm_wmma_f16<false, true, true, false><<<grid512, gblk, 0, stream>>>(
        bufT, wbuf, mlp_b2, nullptr, (float*)d_out, nullptr, B, Dout, Dm);
}
